// LUNAFullQuant_79182017069431
// MI455X (gfx1250) — 16-bit and 8-bit matrix operand order repaired and checked by reading, compiled, not run on hardware
//
#include <hip/hip_runtime.h>
#include <hip/hip_bf16.h>


typedef __attribute__((ext_vector_type(8)))  int      v8i;
typedef __attribute__((ext_vector_type(8)))  float    v8f;
typedef __attribute__((ext_vector_type(16))) _Float16 v16h;
typedef _Float16 f16;

__device__ __forceinline__ float gelu_f(float x) {
    return 0.5f * x * (1.0f + erff(x * 0.70710678118654752f));
}
__device__ __forceinline__ float scale_of(const float* am) {
    return fmaxf(am[0], 1e-8f) * (1.0f / 127.0f);
}
__device__ __forceinline__ float q127(float x, float s) {
    float q = rintf(x / s);
    return fminf(127.0f, fmaxf(-127.0f, q));
}
__device__ __forceinline__ int kpatA8(int v) {
    return ((v & 2) ? 16 : 0) + ((v & 1) ? 4 : 0) + ((v & 4) ? 32 : 0);
}
__device__ __forceinline__ v8i load8i(const int* __restrict__ p) {
    v8i r;
#pragma unroll
    for (int v = 0; v < 8; ++v) r[v] = p[v];
    return r;
}

__global__ void k_absmax(const float* __restrict__ x, size_t n, float* __restrict__ amax) {
    __shared__ float sm[256];
    float m = 0.0f;
    for (size_t i = (size_t)blockIdx.x * blockDim.x + threadIdx.x; i < n;
         i += (size_t)gridDim.x * blockDim.x)
        m = fmaxf(m, fabsf(x[i]));
    sm[threadIdx.x] = m; __syncthreads();
    for (int s = 128; s > 0; s >>= 1) {
        if ((int)threadIdx.x < s) sm[threadIdx.x] = fmaxf(sm[threadIdx.x], sm[threadIdx.x + s]);
        __syncthreads();
    }
    if (threadIdx.x == 0) atomicMax((unsigned int*)amax, __float_as_uint(sm[0]));
}

__global__ void k_fq_f32(const float* __restrict__ x, const float* __restrict__ am,
                         float* __restrict__ y, size_t n) {
    size_t i = (size_t)blockIdx.x * blockDim.x + threadIdx.x;
    if (i >= n) return;
    float s = scale_of(am);
    y[i] = q127(x[i], s) * s;
}

__global__ void k_q8(const float* __restrict__ x, const float* __restrict__ am,
                     int8_t* __restrict__ q, size_t n) {
    size_t i = (size_t)blockIdx.x * blockDim.x + threadIdx.x;
    if (i >= n) return;
    q[i] = (int8_t)q127(x[i], scale_of(am));
}

__global__ void k_q8_sw(const float* __restrict__ x, const float* __restrict__ am,
                        int8_t* __restrict__ q, int M, int Mp, int K, int Kp) {
    size_t i = (size_t)blockIdx.x * blockDim.x + threadIdx.x;
    size_t n = (size_t)Mp * Kp;
    if (i >= n) return;
    int byte = (int)(i & 3); int v = (int)((i >> 2) & 7); int lane = (int)((i >> 5) & 31);
    size_t blk = i >> 10;
    int KB = Kp >> 6;
    int bk = (int)(blk % KB); int bm = (int)(blk / KB);
    int m  = (bm << 4) + (lane & 15);
    int kc = (bk << 6) + kpatA8(v) + ((lane >> 4) << 3) + byte;
    float s = scale_of(am);
    q[i] = (m < M && kc < K) ? (int8_t)q127(x[(size_t)m * K + kc], s) : (int8_t)0;
}

__global__ void k_q8T_sw(const float* __restrict__ w, const float* __restrict__ am,
                         int8_t* __restrict__ qT, int N, int K, int Kp, int Np) {
    size_t i = (size_t)blockIdx.x * blockDim.x + threadIdx.x;
    size_t n = (size_t)Kp * Np;
    if (i >= n) return;
    int byte = (int)(i & 3); int v = (int)((i >> 2) & 7); int lane = (int)((i >> 5) & 31);
    size_t blk = i >> 10;
    int KB = Kp >> 6;
    int bk = (int)(blk % KB); int bn = (int)(blk / KB);
    int kr = (bk << 6) + kpatA8(v) + ((lane >> 4) << 3) + byte;
    int nc = (bn << 4) + (lane & 15);
    float s = scale_of(am);
    qT[i] = (kr < K && nc < N) ? (int8_t)q127(w[(size_t)nc * K + kr], s) : (int8_t)0;
}

__global__ void k_f16_sw(const float* __restrict__ x, f16* __restrict__ y,
                         int M, int Mp, int K, int Kp) {
    size_t j = (size_t)blockIdx.x * blockDim.x + threadIdx.x;
    size_t n = (size_t)Mp * Kp;
    if (j >= n) return;
    int e = (int)(j & 1); int v = (int)((j >> 1) & 7); int lane = (int)((j >> 4) & 31);
    size_t blk = j >> 9;
    int KB = Kp >> 5;
    int bk = (int)(blk % KB); int bm = (int)(blk / KB);
    int m  = (bm << 4) + (lane & 15);
    int kc = (bk << 5) + ((v & 3) << 1) + ((v & 4) ? 16 : 0) + ((lane >> 4) << 3) + e;
    y[j] = (m < M && kc < K) ? (f16)x[(size_t)m * K + kc] : (f16)0.0f;
}

__global__ void k_wf16_sw(const float* __restrict__ w, const float* __restrict__ am,
                          f16* __restrict__ wT, int N, int K, int Kp, int Np) {
    size_t j = (size_t)blockIdx.x * blockDim.x + threadIdx.x;
    size_t n = (size_t)Kp * Np;
    if (j >= n) return;
    int e = (int)(j & 1); int v = (int)((j >> 1) & 7); int lane = (int)((j >> 4) & 31);
    size_t blk = j >> 9;
    int KB = Kp >> 5;
    int bk = (int)(blk % KB); int bn = (int)(blk / KB);
    int kr = (bk << 5) + ((v & 3) << 1) + ((v & 4) ? 16 : 0) + ((lane >> 4) << 3) + e;
    int nc = (bn << 4) + (lane & 15);
    float s = scale_of(am);
    wT[j] = (kr < K && nc < N) ? (f16)(q127(w[(size_t)nc * K + kr], s) * s) : (f16)0.0f;
}

__global__ void k_add(const float* __restrict__ a, const float* __restrict__ b,
                      float* __restrict__ o, size_t n) {
    size_t i = (size_t)blockIdx.x * blockDim.x + threadIdx.x;
    if (i < n) o[i] = a[i] + b[i];
}

__global__ void k_gemm_q8(const int8_t* __restrict__ A, const int8_t* __restrict__ B,
                          const float* __restrict__ amA, const float* __restrict__ amB,
                          const float* __restrict__ bias, float* __restrict__ out,
                          int M, int Mp, int N, int Np, int Kp, int act) {
    int KB   = Kp >> 6;
    int ntN4 = Np >> 6;
    int tile = blockIdx.x * 8 + ((int)threadIdx.x >> 5);
    if (tile >= (Mp >> 4) * ntN4) return;
    int tm = tile / ntN4, tn4 = tile - tm * ntN4;
    int lane = threadIdx.x & 31;
    const int* pa = (const int*)(A + (((size_t)tm * KB) * 32 + lane) * 32);
    const int* pb = (const int*)(B + (((size_t)(tn4 * 4) * KB) * 32 + lane) * 32);
    v8i zero = {0,0,0,0,0,0,0,0};
    v8i acc0 = zero, acc1 = zero, acc2 = zero, acc3 = zero;
    v8i a  = load8i(pa);
    v8i b0 = load8i(pb);
    v8i b1 = load8i(pb + (size_t)1 * KB * 256);
    v8i b2 = load8i(pb + (size_t)2 * KB * 256);
    v8i b3 = load8i(pb + (size_t)3 * KB * 256);
    for (int kb = 1; kb < KB; ++kb) {
        v8i an  = load8i(pa + (size_t)kb * 256);
        v8i bn0 = load8i(pb + ((size_t)0 * KB + kb) * 256);
        v8i bn1 = load8i(pb + ((size_t)1 * KB + kb) * 256);
        v8i bn2 = load8i(pb + ((size_t)2 * KB + kb) * 256);
        v8i bn3 = load8i(pb + ((size_t)3 * KB + kb) * 256);
        acc0 = __builtin_amdgcn_wmma_i32_16x16x64_iu8(true, a, true, b0, acc0, false, false);
        acc1 = __builtin_amdgcn_wmma_i32_16x16x64_iu8(true, a, true, b1, acc1, false, false);
        acc2 = __builtin_amdgcn_wmma_i32_16x16x64_iu8(true, a, true, b2, acc2, false, false);
        acc3 = __builtin_amdgcn_wmma_i32_16x16x64_iu8(true, a, true, b3, acc3, false, false);
        a = an; b0 = bn0; b1 = bn1; b2 = bn2; b3 = bn3;
    }
    acc0 = __builtin_amdgcn_wmma_i32_16x16x64_iu8(true, a, true, b0, acc0, false, false);
    acc1 = __builtin_amdgcn_wmma_i32_16x16x64_iu8(true, a, true, b1, acc1, false, false);
    acc2 = __builtin_amdgcn_wmma_i32_16x16x64_iu8(true, a, true, b2, acc2, false, false);
    acc3 = __builtin_amdgcn_wmma_i32_16x16x64_iu8(true, a, true, b3, acc3, false, false);

    float s = scale_of(amA) * scale_of(amB);
    int mbase = (tm << 4) + ((lane >> 4) << 3);
    int nl = lane & 15;
    v8i accs[4] = {acc0, acc1, acc2, acc3};
#pragma unroll
    for (int j = 0; j < 4; ++j) {
        int ncol = ((tn4 * 4 + j) << 4) + nl;
        if (ncol < N) {
            float bv = bias ? bias[ncol] : 0.0f;
#pragma unroll
            for (int r = 0; r < 8; ++r) {
                int m = mbase + r;
                if (m < M) {
                    float v = s * (float)accs[j][r] + bv;
                    if (act == 1) v = gelu_f(v);
                    out[(size_t)m * N + ncol] = v;
                }
            }
        }
    }
}

__global__ void k_gemm_f16(const f16* __restrict__ A, const f16* __restrict__ B,
                           const float* __restrict__ bias, float* __restrict__ out,
                           int M, int Mp, int N, int Np, int Kp, int act) {
    int KB   = Kp >> 5;
    int ntN4 = Np >> 6;
    int tile = blockIdx.x * 8 + ((int)threadIdx.x >> 5);
    if (tile >= (Mp >> 4) * ntN4) return;
    int tm = tile / ntN4, tn4 = tile - tm * ntN4;
    int lane = threadIdx.x & 31;
    const int* pa = (const int*)(A + (((size_t)tm * KB) * 32 + lane) * 16);
    const int* pb = (const int*)(B + (((size_t)(tn4 * 4) * KB) * 32 + lane) * 16);
    v8f zf8 = {0,0,0,0,0,0,0,0};
    v8f acc0 = zf8, acc1 = zf8, acc2 = zf8, acc3 = zf8;
    union H { v16h h; v8i i; };
    H a, b0, b1, b2, b3;
    a.i  = load8i(pa);
    b0.i = load8i(pb);
    b1.i = load8i(pb + (size_t)1 * KB * 256);
    b2.i = load8i(pb + (size_t)2 * KB * 256);
    b3.i = load8i(pb + (size_t)3 * KB * 256);
    for (int kb = 1; kb < KB; ++kb) {
        H an, bn0, bn1, bn2, bn3;
        an.i  = load8i(pa + (size_t)kb * 256);
        bn0.i = load8i(pb + ((size_t)0 * KB + kb) * 256);
        bn1.i = load8i(pb + ((size_t)1 * KB + kb) * 256);
        bn2.i = load8i(pb + ((size_t)2 * KB + kb) * 256);
        bn3.i = load8i(pb + ((size_t)3 * KB + kb) * 256);
        acc0 = __builtin_amdgcn_wmma_f32_16x16x32_f16(false, a.h, false, b0.h, (short)0,
                                                      acc0, false, false);
        acc1 = __builtin_amdgcn_wmma_f32_16x16x32_f16(false, a.h, false, b1.h, (short)0,
                                                      acc1, false, false);
        acc2 = __builtin_amdgcn_wmma_f32_16x16x32_f16(false, a.h, false, b2.h, (short)0,
                                                      acc2, false, false);
        acc3 = __builtin_amdgcn_wmma_f32_16x16x32_f16(false, a.h, false, b3.h, (short)0,
                                                      acc3, false, false);
        a = an; b0 = bn0; b1 = bn1; b2 = bn2; b3 = bn3;
    }
    acc0 = __builtin_amdgcn_wmma_f32_16x16x32_f16(false, a.h, false, b0.h, (short)0,
                                                  acc0, false, false);
    acc1 = __builtin_amdgcn_wmma_f32_16x16x32_f16(false, a.h, false, b1.h, (short)0,
                                                  acc1, false, false);
    acc2 = __builtin_amdgcn_wmma_f32_16x16x32_f16(false, a.h, false, b2.h, (short)0,
                                                  acc2, false, false);
    acc3 = __builtin_amdgcn_wmma_f32_16x16x32_f16(false, a.h, false, b3.h, (short)0,
                                                  acc3, false, false);

    int mbase = (tm << 4) + ((lane >> 4) << 3);
    int nl = lane & 15;
    v8f accs[4] = {acc0, acc1, acc2, acc3};
#pragma unroll
    for (int j = 0; j < 4; ++j) {
        int ncol = ((tn4 * 4 + j) << 4) + nl;
        if (ncol < N) {
            float bv = bias ? bias[ncol] : 0.0f;
#pragma unroll
            for (int r = 0; r < 8; ++r) {
                int m = mbase + r;
                if (m < M) {
                    float v = accs[j][r] + bv;
                    if (act == 1) v = gelu_f(v);
                    out[(size_t)m * N + ncol] = v;
                }
            }
        }
    }
}

__global__ void k_conv1(const int8_t* __restrict__ xq, const int8_t* __restrict__ wq,
                        const float* amX, const float* amW,
                        const float* __restrict__ cb, float* __restrict__ out) {
    int i = blockIdx.x * blockDim.x + threadIdx.x;
    if (i >= 128 * 16 * 704 * 4) return;
    int ow = i & 3; int q = i >> 2; int t = q % 704; int r = q / 704;
    int oc = r & 15; int b = r >> 4;
    const int8_t* row = xq + ((size_t)b * 704 + t) * 40;
    const int8_t* w   = wq + oc * 19;
    int st = ow * 10 - 9, acc = 0;
#pragma unroll
    for (int k = 0; k < 19; ++k) {
        int iw = st + k;
        if ((unsigned)iw < 40u) acc += (int)row[iw] * (int)w[k];
    }
    out[i] = scale_of(amX) * scale_of(amW) * (float)acc + cb[oc];
}

__global__ void k_conv3x(const int8_t* __restrict__ in, const int8_t* __restrict__ wq,
                         const float* amA, const float* amW,
                         const float* __restrict__ cb, float* __restrict__ out) {
    int i = blockIdx.x * blockDim.x + threadIdx.x;
    if (i >= 128 * 16 * 704 * 4) return;
    int ow = i & 3; int q = i >> 2; int t = q % 704; int r = q / 704;
    int oc = r & 15; int b = r >> 4;
    int acc = 0;
    for (int ic = 0; ic < 16; ++ic) {
        const int8_t* p = in + (((size_t)b * 16 + ic) * 704 + t) * 4;
        const int8_t* w = wq + (oc * 16 + ic) * 3;
#pragma unroll
        for (int k = 0; k < 3; ++k) {
            int iw = ow - 1 + k;
            if ((unsigned)iw < 4u) acc += (int)p[iw] * (int)w[k];
        }
    }
    out[i] = scale_of(amA) * scale_of(amW) * (float)acc + cb[oc];
}

__global__ void k_gn_stat(const float* __restrict__ x, float* __restrict__ stat) {
    int b = blockIdx.x >> 2, g = blockIdx.x & 3;
    const float* base = x + ((size_t)b * 16 + g * 4) * 2816;
    __shared__ float ss[256], sq[256];
    float s = 0.0f, q = 0.0f;
    for (int i = threadIdx.x; i < 4 * 2816; i += 256) { float v = base[i]; s += v; q += v * v; }
    ss[threadIdx.x] = s; sq[threadIdx.x] = q; __syncthreads();
    for (int st = 128; st > 0; st >>= 1) {
        if ((int)threadIdx.x < st) { ss[threadIdx.x] += ss[threadIdx.x + st];
                                     sq[threadIdx.x] += sq[threadIdx.x + st]; }
        __syncthreads();
    }
    if (threadIdx.x == 0) {
        float n = 11264.0f, m = ss[0] / n;
        float var = sq[0] / n - m * m;
        stat[blockIdx.x * 2]     = m;
        stat[blockIdx.x * 2 + 1] = rsqrtf(var + 1e-5f);
    }
}

__global__ void k_gn_apply(float* __restrict__ x, const float* __restrict__ stat,
                           const float* __restrict__ g, const float* __restrict__ bt) {
    int i = blockIdx.x * blockDim.x + threadIdx.x;
    if (i >= 128 * 16 * 2816) return;
    int c = (i / 2816) & 15;
    int b = i / (16 * 2816);
    int bg = b * 4 + (c >> 2);
    float m = stat[bg * 2], rstd = stat[bg * 2 + 1];
    float v = (x[i] - m) * rstd * g[c] + bt[c];
    x[i] = gelu_f(v);
}

__global__ void k_cnn_t(const float* __restrict__ c, float* __restrict__ f) {
    size_t i = (size_t)blockIdx.x * blockDim.x + threadIdx.x;
    if (i >= (size_t)128 * 704 * 64) return;
    int d = (int)(i & 63); size_t r = i >> 6;
    int t = (int)(r % 704); int b = (int)(r / 704);
    int oc = d & 15, ow = d >> 4;
    f[i] = c[(((size_t)b * 16 + oc) * 704 + t) * 4 + ow];
}

__global__ void k_dft(const float* __restrict__ xq, float* __restrict__ feat) {
    int i = blockIdx.x * blockDim.x + threadIdx.x;
    if (i >= 90112 * 21) return;
    int j = i % 21; int r = i / 21;
    const float* p = xq + (size_t)r * 40;
    float re = 0.0f, im = 0.0f;
    float w = -(float)j * (6.283185307179586f / 40.0f);
    for (int n = 0; n < 40; ++n) {
        float sn, cn; __sincosf(w * (float)n, &sn, &cn);
        float v = p[n]; re += v * cn; im += v * sn;
    }
    feat[(size_t)r * 42 + j]      = sqrtf(re * re + im * im);
    feat[(size_t)r * 42 + 21 + j] = atan2f(im, re);
}

__global__ void k_add_nerf(float* __restrict__ tok, const float* __restrict__ nq, size_t n) {
    size_t i = (size_t)blockIdx.x * blockDim.x + threadIdx.x;
    if (i >= n) return;
    int d = (int)(i & 63); size_t r = i >> 6;
    int c = (int)((r % 704) >> 5);
    tok[i] += nq[c * 64 + d];
}

__global__ void k_kv_q8(const float* __restrict__ tok, const float* __restrict__ am,
                        int8_t* __restrict__ kvq, size_t n) {
    size_t i = (size_t)blockIdx.x * blockDim.x + threadIdx.x;
    if (i >= n) return;
    int byte = (int)(i & 3); int v = (int)((i >> 2) & 7); int lane = (int)((i >> 5) & 31);
    int bm = (int)(i >> 10);
    int R  = (bm << 4) + (lane & 15);
    int kc = kpatA8(v) + ((lane >> 4) << 3) + byte;
    int b = R / 704; int rem = R % 704;
    int p = rem / 22; int c = rem - p * 22;
    float s = scale_of(am);
    float val = tok[(((size_t)b * 704) + (size_t)c * 32 + p) * 64 + kc];
    kvq[i] = (int8_t)q127(val, s);
}

__global__ void k_res_ln(const float* __restrict__ x, const float* __restrict__ res,
                         int resMod, const float* __restrict__ g, const float* __restrict__ bt,
                         float* __restrict__ out) {
    extern __shared__ float sm[];
    int D = blockDim.x, row = blockIdx.x, d = threadIdx.x;
    float v = x[(size_t)row * D + d];
    if (res) v += res[(size_t)(resMod ? (row % resMod) : row) * D + d];
    sm[d] = v; __syncthreads();
    for (int st = D >> 1; st > 0; st >>= 1) { if (d < st) sm[d] += sm[d + st]; __syncthreads(); }
    float m = sm[0] / (float)D; __syncthreads();
    float dv = v - m;
    sm[d] = dv * dv; __syncthreads();
    for (int st = D >> 1; st > 0; st >>= 1) { if (d < st) sm[d] += sm[d + st]; __syncthreads(); }
    float rstd = rsqrtf(sm[0] / (float)D + 1e-5f);
    out[(size_t)row * D + d] = dv * rstd * g[d] + bt[d];
}

__global__ void k_attn_ca(const float* __restrict__ qp, const float* __restrict__ kvp,
                          float* __restrict__ out) {
    int grp = blockIdx.x >> 1, h = blockIdx.x & 1;
    __shared__ float s[4][22];
    int tid = threadIdx.x;
    const float* kvb = kvp + (size_t)grp * 22 * 128;
    if (tid < 88) {
        int qi = tid / 22, c = tid % 22;
        const float* qv = qp + qi * 64 + h * 32;
        const float* kv = kvb + c * 128 + h * 32;
        float a = 0.0f;
        for (int d = 0; d < 32; ++d) a += qv[d] * kv[d];
        s[qi][c] = a * 0.17677669529663688f;
    }
    __syncthreads();
    if (tid < 4) {
        float mx = -1e30f; for (int c = 0; c < 22; ++c) mx = fmaxf(mx, s[tid][c]);
        float sum = 0.0f;
        for (int c = 0; c < 22; ++c) { float e = __expf(s[tid][c] - mx); s[tid][c] = e; sum += e; }
        float inv = 1.0f / sum;
        for (int c = 0; c < 22; ++c) s[tid][c] *= inv;
    }
    __syncthreads();
    int qi = tid >> 5, d = tid & 31;
    const float* vv = kvb + 64 + h * 32 + d;
    float a = 0.0f;
    for (int c = 0; c < 22; ++c) a += s[qi][c] * vv[c * 128];
    out[((size_t)grp * 4 + qi) * 64 + h * 32 + d] = a;
}

__global__ void k_attn_sa(const float* __restrict__ qkv, float* __restrict__ out) {
    int grp = blockIdx.x >> 1, h = blockIdx.x & 1;
    __shared__ float s[4][4];
    int tid = threadIdx.x;
    const float* base = qkv + (size_t)grp * 4 * 192;
    if (tid < 16) {
        int qi = tid >> 2, ki = tid & 3;
        const float* qv = base + qi * 192 + h * 32;
        const float* kv = base + ki * 192 + 64 + h * 32;
        float a = 0.0f;
        for (int d = 0; d < 32; ++d) a += qv[d] * kv[d];
        s[qi][ki] = a * 0.17677669529663688f;
    }
    __syncthreads();
    if (tid < 4) {
        float mx = -1e30f; for (int k = 0; k < 4; ++k) mx = fmaxf(mx, s[tid][k]);
        float sum = 0.0f;
        for (int k = 0; k < 4; ++k) { float e = __expf(s[tid][k] - mx); s[tid][k] = e; sum += e; }
        float inv = 1.0f / sum;
        for (int k = 0; k < 4; ++k) s[tid][k] *= inv;
    }
    __syncthreads();
    int qi = tid >> 5, d = tid & 31;
    float a = 0.0f;
    for (int k = 0; k < 4; ++k) a += s[qi][k] * base[k * 192 + 128 + h * 32 + d];
    out[((size_t)grp * 4 + qi) * 64 + h * 32 + d] = a;
}

__global__ void k_attn_blk(const float* __restrict__ qkv, float* __restrict__ out) {
    int b = blockIdx.x >> 3, h = blockIdx.x & 7;
    __shared__ float qs[32][33], ks[32][33], s[32][33];
    int tid = threadIdx.x;
    for (int i = tid; i < 1024; i += 256) {
        int p = i >> 5, d = i & 31;
        const float* row = qkv + ((size_t)b * 32 + p) * 768 + h * 32;
        int i0 = d & 15;
        float inv = __expf(-(float)i0 * (9.210340371976184f / 16.0f));
        float sn, cn; __sincosf((float)p * inv, &sn, &cn);
        float q1 = row[i0],       q2 = row[i0 + 16];
        float k1 = row[256 + i0], k2 = row[256 + i0 + 16];
        if (d < 16) { qs[p][d] = q1 * cn - q2 * sn; ks[p][d] = k1 * cn - k2 * sn; }
        else        { qs[p][d] = q1 * sn + q2 * cn; ks[p][d] = k1 * sn + k2 * cn; }
    }
    __syncthreads();
    for (int i = tid; i < 1024; i += 256) {
        int p = i >> 5, k = i & 31;
        float a = 0.0f;
        for (int d = 0; d < 32; ++d) a += qs[p][d] * ks[k][d];
        s[p][k] = a * 0.17677669529663688f;
    }
    __syncthreads();
    if (tid < 32) {
        float mx = -1e30f; for (int k = 0; k < 32; ++k) mx = fmaxf(mx, s[tid][k]);
        float sum = 0.0f;
        for (int k = 0; k < 32; ++k) { float e = __expf(s[tid][k] - mx); s[tid][k] = e; sum += e; }
        float inv = 1.0f / sum;
        for (int k = 0; k < 32; ++k) s[tid][k] *= inv;
    }
    __syncthreads();
    for (int i = tid; i < 1024; i += 256) {
        int p = i >> 5, d = i & 31;
        float a = 0.0f;
        for (int k = 0; k < 32; ++k)
            a += s[p][k] * qkv[((size_t)b * 32 + k) * 768 + 512 + h * 32 + d];
        out[((size_t)b * 32 + p) * 256 + h * 32 + d] = a;
    }
}

__global__ void k_attn_cls(const float* __restrict__ qp, const float* __restrict__ kvp,
                           float* __restrict__ out) {
    int b = blockIdx.x >> 3, h = blockIdx.x & 7;
    __shared__ float s[32];
    int tid = threadIdx.x;
    const float* q   = qp + h * 32;
    const float* kvb = kvp + (size_t)b * 32 * 512;
    {
        const float* kv = kvb + tid * 512 + h * 32;
        float a = 0.0f;
        for (int d = 0; d < 32; ++d) a += q[d] * kv[d];
        s[tid] = a * 0.17677669529663688f;
    }
    __syncthreads();
    if (tid == 0) {
        float mx = -1e30f; for (int k = 0; k < 32; ++k) mx = fmaxf(mx, s[k]);
        float sum = 0.0f;
        for (int k = 0; k < 32; ++k) { float e = __expf(s[k] - mx); s[k] = e; sum += e; }
        float inv = 1.0f / sum;
        for (int k = 0; k < 32; ++k) s[k] *= inv;
    }
    __syncthreads();
    float a = 0.0f;
    for (int k = 0; k < 32; ++k) a += s[k] * kvb[k * 512 + 256 + h * 32 + tid];
    out[(size_t)b * 256 + h * 32 + tid] = a;
}

static inline size_t cdivs(size_t a, size_t b) { return (a + b - 1) / b; }

extern "C" void kernel_launch(void* const* d_in, const int* in_sizes, int n_in,
                              void* d_out, int out_size, void* d_ws, size_t ws_size,
                              hipStream_t stream) {
    (void)in_sizes; (void)n_in; (void)out_size; (void)ws_size;
    const float* x        = (const float*)d_in[0];
    const float* conv1_w  = (const float*)d_in[1];  const float* conv1_b = (const float*)d_in[2];
    const float* gn1_g    = (const float*)d_in[3];  const float* gn1_b   = (const float*)d_in[4];
    const float* conv2_w  = (const float*)d_in[5];  const float* conv2_b = (const float*)d_in[6];
    const float* gn2_g    = (const float*)d_in[7];  const float* gn2_b   = (const float*)d_in[8];
    const float* conv3_w  = (const float*)d_in[9];  const float* conv3_b = (const float*)d_in[10];
    const float* gn3_g    = (const float*)d_in[11]; const float* gn3_b   = (const float*)d_in[12];
    const float* rfft_w1  = (const float*)d_in[13]; const float* rfft_b1 = (const float*)d_in[14];
    const float* rfft_w2  = (const float*)d_in[15]; const float* rfft_b2 = (const float*)d_in[16];
    const float* nerf_embed=(const float*)d_in[17];
    const float* ca_queries=(const float*)d_in[18];
    const float* ca_wq    = (const float*)d_in[19]; const float* ca_bq   = (const float*)d_in[20];
    const float* ca_wkv   = (const float*)d_in[21]; const float* ca_bkv  = (const float*)d_in[22];
    const float* ca_wo    = (const float*)d_in[23]; const float* ca_bo   = (const float*)d_in[24];
    const float* ca_ln1_g = (const float*)d_in[25]; const float* ca_ln1_b= (const float*)d_in[26];
    const float* sa_wqkv  = (const float*)d_in[27]; const float* sa_bqkv = (const float*)d_in[28];
    const float* sa_wo    = (const float*)d_in[29]; const float* sa_bo   = (const float*)d_in[30];
    const float* ca_ln2_g = (const float*)d_in[31]; const float* ca_ln2_b= (const float*)d_in[32];
    const float* ff_w1    = (const float*)d_in[33]; const float* ff_b1   = (const float*)d_in[34];
    const float* ff_w2    = (const float*)d_in[35]; const float* ff_b2   = (const float*)d_in[36];
    const float* ca_ln3_g = (const float*)d_in[37]; const float* ca_ln3_b= (const float*)d_in[38];
    const float* blk_ln1_g= (const float*)d_in[39]; const float* blk_ln1_b=(const float*)d_in[40];
    const float* blk_wqkv = (const float*)d_in[41]; const float* blk_bqkv= (const float*)d_in[42];
    const float* blk_wo   = (const float*)d_in[43]; const float* blk_bo  = (const float*)d_in[44];
    const float* blk_ln2_g= (const float*)d_in[45]; const float* blk_ln2_b=(const float*)d_in[46];
    const float* blk_w1   = (const float*)d_in[47]; const float* blk_b1  = (const float*)d_in[48];
    const float* blk_w2   = (const float*)d_in[49]; const float* blk_b2  = (const float*)d_in[50];
    const float* fin_g    = (const float*)d_in[51]; const float* fin_b   = (const float*)d_in[52];
    const float* cls_q    = (const float*)d_in[53];
    const float* cls_wq   = (const float*)d_in[54]; const float* cls_bq  = (const float*)d_in[55];
    const float* cls_wkv  = (const float*)d_in[56]; const float* cls_bkv = (const float*)d_in[57];
    const float* cls_wo   = (const float*)d_in[58]; const float* cls_bo  = (const float*)d_in[59];
    const float* head_w1  = (const float*)d_in[60]; const float* head_b1 = (const float*)d_in[61];
    const float* head_w2  = (const float*)d_in[62]; const float* head_b2 = (const float*)d_in[63];

    const int B = 128, NT = 704, NTOK = B * NT;
    const int G16 = 16384;
    const int BN = 4096;

    size_t off = 0;
    auto alloc = [&](size_t bytes) -> char* {
        off = (off + 255) & ~(size_t)255;
        char* p = (char*)d_ws + off; off += bytes; return p;
    };
    float*  amax   = (float*)alloc(512 * 4);
    int8_t* c1wq   = (int8_t*)alloc(16 * 19);
    int8_t* c2wq   = (int8_t*)alloc(16 * 16 * 3);
    int8_t* c3wq   = (int8_t*)alloc(16 * 16 * 3);
    int8_t* rw1T   = (int8_t*)alloc(64 * 192);
    int8_t* rw2T   = (int8_t*)alloc(192 * 64);
    int8_t* cawqT  = (int8_t*)alloc(64 * 64);
    int8_t* cawkvT = (int8_t*)alloc(64 * 128);
    int8_t* sawqkvT= (int8_t*)alloc(64 * 192);
    int8_t* ffw1T  = (int8_t*)alloc(64 * 256);
    int8_t* ffw2T  = (int8_t*)alloc(256 * 64);
    int8_t* clswqT = (int8_t*)alloc(256 * 256);
    int8_t* clswkvT= (int8_t*)alloc(256 * 512);
    int8_t* hw1T   = (int8_t*)alloc(256 * 64);
    int8_t* hw2T   = (int8_t*)alloc(64 * 64);
    int8_t* bwqkvT = (int8_t*)alloc((size_t)8 * 256 * 768);
    int8_t* bw1T   = (int8_t*)alloc((size_t)8 * 256 * 1024);
    int8_t* bw2T   = (int8_t*)alloc((size_t)8 * 1024 * 256);
    f16*    cawoT  = (f16*)alloc(64 * 64 * 2);
    f16*    sawoT  = (f16*)alloc(64 * 64 * 2);
    f16*    clswoT = (f16*)alloc(256 * 256 * 2);
    f16*    bwoT   = (f16*)alloc((size_t)8 * 256 * 256 * 2);
    float*  nerfq  = (float*)alloc(22 * 64 * 4);
    int8_t* caq8   = (int8_t*)alloc(16 * 64);
    int8_t* clsq8  = (int8_t*)alloc(16 * 256);
    float*  qp_ca  = (float*)alloc(4 * 64 * 4);
    float*  qp_cls = (float*)alloc(256 * 4);
    float*  gstat  = (float*)alloc(1024 * 4);
    float*  z      = (float*)alloc((size_t)BN * 256 * 4);
    float*  xqF    = (float*)alloc((size_t)NTOK * 40 * 4);
    int8_t* xqQ    = (int8_t*)alloc((size_t)NTOK * 40);
    float*  featA  = (float*)alloc((size_t)NTOK * 42 * 4);
    int8_t* fq8A   = (int8_t*)alloc((size_t)NTOK * 64);
    float*  c1buf  = (float*)alloc((size_t)B * 16 * NT * 4 * 4);
    float*  c2buf  = (float*)alloc((size_t)B * 16 * NT * 4 * 4);
    int8_t* cqbuf  = (int8_t*)alloc((size_t)B * 16 * NT * 4);
    float*  h2A    = (float*)alloc((size_t)NTOK * 168 * 4);
    int8_t* h2qA   = (int8_t*)alloc((size_t)NTOK * 192);

    float*  feat   = featA;
    float*  u0     = featA;
    f16*    u0h    = (f16*)(featA + 1048576);
    float*  wo1    = featA + 1572864;
    float*  u1     = featA + 2621440;
    float*  u2     = xqF;
    int8_t* u1q    = (int8_t*)(xqF + 1048576);
    int8_t* u2q    = (int8_t*)(xqF + 1310720);
    float*  h2     = h2A;
    float*  h2b    = h2A;
    float*  cnn    = c2buf;
    float*  tok    = c1buf;
    int8_t* featq  = fq8A;
    int8_t* kvq    = fq8A;
    float*  kvp    = h2A;
    float*  qkvsa  = h2A;
    float*  sa_o   = h2A + 3145728;
    f16*    sa_oh  = (f16*)(h2A + 4194304);
    float*  wo2    = h2A + 4718592;
    float*  ffmca  = h2A + 5767168;
    int8_t* ffmcq  = (int8_t*)(h2A + 9961472);
    float*  ff2ca  = h2A + 11010048;
    float*  hh     = h2A;
    int8_t* hhq    = (int8_t*)(h2A + 1048576);
    float*  qkvb   = h2A + 1310720;
    float*  ob     = h2A + 4456448;
    f16*    obh    = (f16*)(h2A + 5505024);
    float*  wob    = h2A + 6029312;
    float*  ffm    = h2A + 7077888;
    int8_t* ffmq   = (int8_t*)(h2A + 11272192);
    float*  ff2    = h2A + 12320768;
    float*  zf     = h2A;
    int8_t* zfq    = (int8_t*)(h2A + 1048576);
    float*  kvpc   = h2A + 1310720;
    float*  poolA  = h2A + 3407872;
    f16*    poolAh = (f16*)(h2A + 3440640);
    float*  pooled = h2A + 3457024;
    int8_t* poolq  = (int8_t*)(h2A + 3489792);
    float*  hc     = h2A + 3497984;
    int8_t* hcq    = (int8_t*)(h2A + 3506176);

    int ns = 0; auto S = [&]() { return ns++; };
    int s_x=S(), s_c1w=S(), s_c2w=S(), s_c3w=S(), s_rw1=S(), s_rw2=S(), s_cawq=S(),
        s_cawkv=S(), s_sawqkv=S(), s_ffw1=S(), s_ffw2=S(), s_clswq=S(), s_clswkv=S(),
        s_hw1=S(), s_hw2=S(), s_cawo=S(), s_sawo=S(), s_clswo=S(), s_nerf=S(),
        s_caq=S(), s_clsq=S();
    int s_bwqkv0 = ns; ns += 8; int s_bw1_0 = ns; ns += 8;
    int s_bw2_0  = ns; ns += 8; int s_bwo0  = ns; ns += 8;
    int s_g1=S(), s_g2=S(), s_g3=S(), s_feat=S(), s_h2=S(), s_h2b=S(), s_tok1=S(),
        s_tok=S(), s_u1=S(), s_u2=S(), s_ffca=S(), s_z0=S();
    int s_blk0 = ns; ns += 8 * 6;
    int s_zf=S(), s_pooled=S(), s_hc=S();
    (void)s_z0;

    hipMemsetAsync(amax, 0, 512 * sizeof(float), stream);

    auto absmax = [&](const float* p, size_t n, int slot) {
        size_t blk = cdivs(n, 256); if (blk > 4096) blk = 4096;
        k_absmax<<<dim3((unsigned)blk), dim3(256), 0, stream>>>(p, n, amax + slot);
    };
    auto fqf = [&](float* p, size_t n, int slot) {
        absmax(p, n, slot);
        k_fq_f32<<<dim3((unsigned)cdivs(n, 256)), dim3(256), 0, stream>>>(p, amax + slot, p, n);
    };
    auto q8sw = [&](const float* p, int M, int Mp, int K, int Kp, int slot, int8_t* dst) {
        size_t n = (size_t)Mp * Kp;
        k_q8_sw<<<dim3((unsigned)cdivs(n, 256)), dim3(256), 0, stream>>>(
            p, amax + slot, dst, M, Mp, K, Kp);
    };
    auto quantW8 = [&](const float* w, int N, int K, int Kp, int Np, int slot, int8_t* dst) {
        absmax(w, (size_t)N * K, slot);
        size_t n = (size_t)Kp * Np;
        k_q8T_sw<<<dim3((unsigned)cdivs(n, 256)), dim3(256), 0, stream>>>(
            w, amax + slot, dst, N, K, Kp, Np);
    };
    auto quantW16 = [&](const float* w, int N, int K, int Kp, int Np, int slot, f16* dst) {
        absmax(w, (size_t)N * K, slot);
        size_t n = (size_t)Kp * Np;
        k_wf16_sw<<<dim3((unsigned)cdivs(n, 256)), dim3(256), 0, stream>>>(
            w, amax + slot, dst, N, K, Kp, Np);
    };
    auto f16sw = [&](const float* p, f16* dst, int M, int Mp, int K, int Kp) {
        size_t n = (size_t)Mp * Kp;
        k_f16_sw<<<dim3((unsigned)cdivs(n, 256)), dim3(256), 0, stream>>>(p, dst, M, Mp, K, Kp);
    };
    auto gemm8 = [&](const int8_t* A, const int8_t* Bw, int sa, int sb, const float* bias,
                     float* outp, int M, int Mp, int N, int Np, int Kp, int act) {
        size_t tiles = (size_t)(Mp / 16) * (Np / 64);
        k_gemm_q8<<<dim3((unsigned)cdivs(tiles, 8)), dim3(256), 0, stream>>>(
            A, Bw, amax + sa, amax + sb, bias, outp, M, Mp, N, Np, Kp, act);
    };
    auto gemmH = [&](const f16* A, const f16* Bw, const float* bias, float* outp,
                     int M, int Mp, int N, int Np, int Kp, int act) {
        size_t tiles = (size_t)(Mp / 16) * (Np / 64);
        k_gemm_f16<<<dim3((unsigned)cdivs(tiles, 8)), dim3(256), 0, stream>>>(
            A, Bw, bias, outp, M, Mp, N, Np, Kp, act);
    };
    auto ew = [&](size_t n) { return dim3((unsigned)cdivs(n, 256)); };

    absmax(conv1_w, 16 * 19, s_c1w);
    k_q8<<<ew(304), 256, 0, stream>>>(conv1_w, amax + s_c1w, c1wq, 304);
    absmax(conv2_w, 768, s_c2w);
    k_q8<<<ew(768), 256, 0, stream>>>(conv2_w, amax + s_c2w, c2wq, 768);
    absmax(conv3_w, 768, s_c3w);
    k_q8<<<ew(768), 256, 0, stream>>>(conv3_w, amax + s_c3w, c3wq, 768);
    quantW8(rfft_w1, 168, 42, 64, 192, s_rw1, rw1T);
    quantW8(rfft_w2, 64, 168, 192, 64, s_rw2, rw2T);
    quantW8(ca_wq, 64, 64, 64, 64, s_cawq, cawqT);
    quantW8(ca_wkv, 128, 64, 64, 128, s_cawkv, cawkvT);
    quantW8(sa_wqkv, 192, 64, 64, 192, s_sawqkv, sawqkvT);
    quantW8(ff_w1, 256, 64, 64, 256, s_ffw1, ffw1T);
    quantW8(ff_w2, 64, 256, 256, 64, s_ffw2, ffw2T);
    quantW8(cls_wq, 256, 256, 256, 256, s_clswq, clswqT);
    quantW8(cls_wkv, 512, 256, 256, 512, s_clswkv, clswkvT);
    quantW8(head_w1, 64, 256, 256, 64, s_hw1, hw1T);
    quantW8(head_w2, 2, 64, 64, 64, s_hw2, hw2T);
    quantW16(ca_wo, 64, 64, 64, 64, s_cawo, cawoT);
    quantW16(sa_wo, 64, 64, 64, 64, s_sawo, sawoT);
    quantW16(cls_wo, 256, 256, 256, 256, s_clswo, clswoT);
    for (int d = 0; d < 8; ++d) {
        quantW8(blk_wqkv + (size_t)d * 768 * 256, 768, 256, 256, 768, s_bwqkv0 + d,
                bwqkvT + (size_t)d * 256 * 768);
        quantW8(blk_w1 + (size_t)d * 1024 * 256, 1024, 256, 256, 1024, s_bw1_0 + d,
                bw1T + (size_t)d * 256 * 1024);
        quantW8(blk_w2 + (size_t)d * 256 * 1024, 256, 1024, 1024, 256, s_bw2_0 + d,
                bw2T + (size_t)d * 1024 * 256);
        quantW16(blk_wo + (size_t)d * 256 * 256, 256, 256, 256, 256, s_bwo0 + d,
                 bwoT + (size_t)d * 256 * 256);
    }
    absmax(nerf_embed, 22 * 64, s_nerf);
    k_fq_f32<<<ew(22 * 64), 256, 0, stream>>>(nerf_embed, amax + s_nerf, nerfq, 22 * 64);
    absmax(ca_queries, 4 * 64, s_caq);
    q8sw(ca_queries, 4, 16, 64, 64, s_caq, caq8);
    absmax(cls_q, 256, s_clsq);
    q8sw(cls_q, 1, 16, 256, 256, s_clsq, clsq8);

    const size_t NX = (size_t)NTOK * 40;
    absmax(x, NX, s_x);
    k_fq_f32<<<ew(NX), 256, 0, stream>>>(x, amax + s_x, xqF, NX);
    k_q8<<<ew(NX), 256, 0, stream>>>(x, amax + s_x, xqQ, NX);

    k_dft<<<ew((size_t)NTOK * 21), 256, 0, stream>>>(xqF, feat);

    const size_t NC = (size_t)B * 16 * NT * 4;
    k_conv1<<<ew(NC), 256, 0, stream>>>(xqQ, c1wq, amax + s_x, amax + s_c1w, conv1_b, c1buf);
    k_gn_stat<<<512, 256, 0, stream>>>(c1buf, gstat);
    k_gn_apply<<<ew(NC), 256, 0, stream>>>(c1buf, gstat, gn1_g, gn1_b);
    absmax(c1buf, NC, s_g1);
    k_q8<<<ew(NC), 256, 0, stream>>>(c1buf, amax + s_g1, cqbuf, NC);
    k_conv3x<<<ew(NC), 256, 0, stream>>>(cqbuf, c2wq, amax + s_g1, amax + s_c2w, conv2_b, c2buf);
    k_gn_stat<<<512, 256, 0, stream>>>(c2buf, gstat);
    k_gn_apply<<<ew(NC), 256, 0, stream>>>(c2buf, gstat, gn2_g, gn2_b);
    absmax(c2buf, NC, s_g2);
    k_q8<<<ew(NC), 256, 0, stream>>>(c2buf, amax + s_g2, cqbuf, NC);
    k_conv3x<<<ew(NC), 256, 0, stream>>>(cqbuf, c3wq, amax + s_g2, amax + s_c3w, conv3_b, c1buf);
    k_gn_stat<<<512, 256, 0, stream>>>(c1buf, gstat);
    k_gn_apply<<<ew(NC), 256, 0, stream>>>(c1buf, gstat, gn3_g, gn3_b);
    fqf(c1buf, NC, s_g3);
    k_cnn_t<<<ew((size_t)NTOK * 64), 256, 0, stream>>>(c1buf, cnn);

    absmax(feat, (size_t)NTOK * 42, s_feat);
    q8sw(feat, NTOK, NTOK, 42, 64, s_feat, featq);
    gemm8(featq, rw1T, s_feat, s_rw1, rfft_b1, h2, NTOK, NTOK, 168, 192, 64, 1);
    absmax(h2, (size_t)NTOK * 168, s_h2);
    q8sw(h2, NTOK, NTOK, 168, 192, s_h2, h2qA);
    gemm8(h2qA, rw2T, s_h2, s_rw2, rfft_b2, h2b, NTOK, NTOK, 64, 64, 192, 0);
    fqf(h2b, (size_t)NTOK * 64, s_h2b);

    k_add<<<ew((size_t)NTOK * 64), 256, 0, stream>>>(cnn, h2b, tok, (size_t)NTOK * 64);
    fqf(tok, (size_t)NTOK * 64, s_tok1);
    k_add_nerf<<<ew((size_t)NTOK * 64), 256, 0, stream>>>(tok, nerfq, (size_t)NTOK * 64);
    fqf(tok, (size_t)NTOK * 64, s_tok);

    k_kv_q8<<<ew((size_t)NTOK * 64), 256, 0, stream>>>(tok, amax + s_tok, kvq,
                                                       (size_t)NTOK * 64);
    gemm8(kvq, cawkvT, s_tok, s_cawkv, ca_bkv, kvp, NTOK, NTOK, 128, 128, 64, 0);
    gemm8(caq8, cawqT, s_caq, s_cawq, ca_bq, qp_ca, 4, 16, 64, 64, 64, 0);
    k_attn_ca<<<8192, 128, 0, stream>>>(qp_ca, kvp, u0);
    f16sw(u0, u0h, G16, G16, 64, 64);
    gemmH(u0h, cawoT, ca_bo, wo1, G16, G16, 64, 64, 64, 0);
    k_res_ln<<<G16, 64, 64 * 4, stream>>>(wo1, ca_queries, 4, ca_ln1_g, ca_ln1_b, u1);
    fqf(u1, (size_t)G16 * 64, s_u1);
    q8sw(u1, G16, G16, 64, 64, s_u1, u1q);
    gemm8(u1q, sawqkvT, s_u1, s_sawqkv, sa_bqkv, qkvsa, G16, G16, 192, 192, 64, 0);
    k_attn_sa<<<8192, 128, 0, stream>>>(qkvsa, sa_o);
    f16sw(sa_o, sa_oh, G16, G16, 64, 64);
    gemmH(sa_oh, sawoT, sa_bo, wo2, G16, G16, 64, 64, 64, 0);
    k_res_ln<<<G16, 64, 64 * 4, stream>>>(wo2, u1, 0, ca_ln2_g, ca_ln2_b, u2);
    fqf(u2, (size_t)G16 * 64, s_u2);
    q8sw(u2, G16, G16, 64, 64, s_u2, u2q);
    gemm8(u2q, ffw1T, s_u2, s_ffw1, ff_b1, ffmca, G16, G16, 256, 256, 64, 1);
    absmax(ffmca, (size_t)G16 * 256, s_ffca);
    q8sw(ffmca, G16, G16, 256, 256, s_ffca, ffmcq);
    gemm8(ffmcq, ffw2T, s_ffca, s_ffw2, ff_b2, ff2ca, G16, G16, 64, 64, 256, 0);
    k_res_ln<<<G16, 64, 64 * 4, stream>>>(ff2ca, u2, 0, ca_ln3_g, ca_ln3_b, z);
    fqf(z, (size_t)BN * 256, s_z0);

    const size_t NZ = (size_t)BN * 256;
    for (int d = 0; d < 8; ++d) {
        int sb = s_blk0 + d * 6;
        int s_hh1 = sb + 0, s_wo = sb + 1, s_z1 = sb + 2, s_hh2 = sb + 3,
            s_fm = sb + 4, s_z2 = sb + 5;
        k_res_ln<<<BN, 256, 256 * 4, stream>>>(z, nullptr, 0, blk_ln1_g + d * 256,
                                               blk_ln1_b + d * 256, hh);
        absmax(hh, NZ, s_hh1);
        q8sw(hh, BN, BN, 256, 256, s_hh1, hhq);
        gemm8(hhq, bwqkvT + (size_t)d * 256 * 768, s_hh1, s_bwqkv0 + d,
              blk_bqkv + d * 768, qkvb, BN, BN, 768, 768, 256, 0);
        k_attn_blk<<<1024, 256, 0, stream>>>(qkvb, ob);
        f16sw(ob, obh, BN, BN, 256, 256);
        gemmH(obh, bwoT + (size_t)d * 256 * 256, blk_bo + d * 256, wob,
              BN, BN, 256, 256, 256, 0);
        fqf(wob, NZ, s_wo);
        k_add<<<ew(NZ), 256, 0, stream>>>(wob, z, z, NZ);
        fqf(z, NZ, s_z1);
        k_res_ln<<<BN, 256, 256 * 4, stream>>>(z, nullptr, 0, blk_ln2_g + d * 256,
                                               blk_ln2_b + d * 256, hh);
        absmax(hh, NZ, s_hh2);
        q8sw(hh, BN, BN, 256, 256, s_hh2, hhq);
        gemm8(hhq, bw1T + (size_t)d * 256 * 1024, s_hh2, s_bw1_0 + d,
              blk_b1 + d * 1024, ffm, BN, BN, 1024, 1024, 256, 1);
        absmax(ffm, (size_t)BN * 1024, s_fm);
        q8sw(ffm, BN, BN, 1024, 1024, s_fm, ffmq);
        gemm8(ffmq, bw2T + (size_t)d * 1024 * 256, s_fm, s_bw2_0 + d,
              blk_b2 + d * 256, ff2, BN, BN, 256, 256, 1024, 0);
        k_add<<<ew(NZ), 256, 0, stream>>>(ff2, z, z, NZ);
        fqf(z, NZ, s_z2);
    }

    k_res_ln<<<BN, 256, 256 * 4, stream>>>(z, nullptr, 0, fin_g, fin_b, zf);
    absmax(zf, NZ, s_zf);
    q8sw(zf, BN, BN, 256, 256, s_zf, zfq);
    gemm8(clsq8, clswqT, s_clsq, s_clswq, cls_bq, qp_cls, 1, 16, 256, 256, 256, 0);
    gemm8(zfq, clswkvT, s_zf, s_clswkv, cls_bkv, kvpc, BN, BN, 512, 512, 256, 0);
    k_attn_cls<<<1024, 32, 0, stream>>>(qp_cls, kvpc, poolA);
    f16sw(poolA, poolAh, B, B, 256, 256);
    gemmH(poolAh, clswoT, cls_bo, pooled, B, B, 256, 256, 256, 0);
    absmax(pooled, (size_t)B * 256, s_pooled);
    q8sw(pooled, B, B, 256, 256, s_pooled, poolq);
    gemm8(poolq, hw1T, s_pooled, s_hw1, head_b1, hc, B, B, 64, 64, 256, 1);
    absmax(hc, (size_t)B * 64, s_hc);
    q8sw(hc, B, B, 64, 64, s_hc, hcq);
    gemm8(hcq, hw2T, s_hc, s_hw2, head_b2, (float*)d_out, B, B, 2, 64, 64, 0);
}
